// Attention_13365938226009
// MI455X (gfx1250) — matrix operand order repaired and checked by reading, not compiled, not run on hardware
//
#include <hip/hip_runtime.h>

// ---------------------------------------------------------------------------
// MI455X (gfx1250) fused attention, bf16 WMMA, flash-style online softmax.
//
// Shapes: B=8, N=1024, C=256, H=8, D=32, S=256 (bias tiled 4x4).
// Total ~13 GFLOP, working set ~25MB (fits 192MB L2). Avoid materializing the
// 256MB score tensor via online softmax; all matmuls on v_wmma_f32_16x16x32_bf16
// (K=32 == head dim, so one WMMA = one full 16x16 score tile).
// ---------------------------------------------------------------------------

typedef __bf16 bf16;
typedef bf16  v16bf __attribute__((ext_vector_type(16)));
typedef bf16  v8bf  __attribute__((ext_vector_type(8)));
typedef float v8f   __attribute__((ext_vector_type(8)));

#define LOG2E 1.4426950408889634f

#define Bsz 8
#define Nn  1024
#define Cc  256
#define Hh  8
#define Dd  32
#define Ss  256

__device__ __forceinline__ int lane_id() { return (int)(threadIdx.x & 31); }

// A fragment (16x32 bf16, M x K), row-major source with leading dim `ld`.
// Lane l holds row m = l&15; lanes 0-15 K in {0..15},
// lanes 16-31 K in {16..31}; one contiguous 32-byte load per lane.
__device__ __forceinline__ v16bf load_frag_a(const bf16* base, int ld) {
  int l  = lane_id();
  int m  = l & 15;
  int kb = (l & 16) ? 16 : 0;
  return *(const v16bf*)(base + m * ld + kb);
}

// B fragment (32x16 bf16, K x N) read from COLUMN-MAJOR storage (i.e. baseT is
// B^T row-major, ld = its leading dim). Lane l: column n = l&15, K-half by
// l>=16; one contiguous 32-byte load per lane.
__device__ __forceinline__ v16bf load_frag_bT(const bf16* baseT, int ld) {
  int l  = lane_id();
  int n  = l & 15;
  int kb = (l & 16) ? 16 : 0;
  return *(const v16bf*)(baseT + n * ld + kb);
}

__device__ __forceinline__ v8f wmma_bf16(v16bf a, v16bf b, v8f c) {
  return __builtin_amdgcn_wmma_f32_16x16x32_bf16(
      /*neg_a=*/false, a, /*neg_b=*/false, b,
      /*c_mod=*/(short)0, c, /*reuse_a=*/false, /*reuse_b=*/false);
}

// ---------------------------------------------------------------------------
// Conversion kernels
// ---------------------------------------------------------------------------
__global__ void cvt_f32_to_bf16(const float* __restrict__ src,
                                bf16* __restrict__ dst, int n) {
  int i = blockIdx.x * blockDim.x + threadIdx.x;
  int stride = gridDim.x * blockDim.x;
  for (; i < n; i += stride) dst[i] = (bf16)src[i];
}

// W [256x256] (k,c) row-major -> Wt (c,k) bf16, optionally pre-scaled.
__global__ void cvt_w_transpose(const float* __restrict__ W,
                                bf16* __restrict__ Wt, float scale) {
  int i = blockIdx.x * blockDim.x + threadIdx.x;  // 65536 threads
  int k = i >> 8, c = i & 255;
  Wt[c * Cc + k] = (bf16)(W[k * Cc + c] * scale);
}

// ---------------------------------------------------------------------------
// Fused QKV projection GEMM: [1024x256] @ [256x256] per (batch, weight).
// widx 0 -> Q ([b][h][n][d], pre-scaled via Wt), 1 -> K ([b][h][n][d]),
// 2 -> V stored transposed Vt ([b][h][d][n]) for contiguous PV B-fragments.
// Block = 8 waves; wave computes 16 rows x 256 cols (16 C fragments).
// ---------------------------------------------------------------------------
__global__ __launch_bounds__(256) void qkv_gemm(
    const bf16* __restrict__ xb, const bf16* __restrict__ Wtq,
    const bf16* __restrict__ Wtk, const bf16* __restrict__ Wtv,
    bf16* __restrict__ Q, bf16* __restrict__ K, bf16* __restrict__ Vt) {
  int blk  = blockIdx.x;
  int widx = blk % 3;
  int rb   = (blk / 3) & 7;
  int b    = blk / 24;
  int w    = threadIdx.x >> 5;
  int m0   = rb * 128 + w * 16;  // row within batch b

  const bf16* Wt = (widx == 0) ? Wtq : (widx == 1) ? Wtk : Wtv;

  v8f acc[16] = {};
  const bf16* Abase = xb + (b * Nn + m0) * Cc;
  for (int kk = 0; kk < Cc; kk += 32) {
    v16bf a = load_frag_a(Abase + kk, Cc);
#pragma unroll
    for (int g = 0; g < 16; ++g) {
      v16bf bfrag = load_frag_bT(Wt + (g * 16) * Cc + kk, Cc);
      acc[g] = wmma_bf16(a, bfrag, acc[g]);
    }
  }

  int l = lane_id();
  int n = l & 15, mb = (l & 16) ? 8 : 0;
#pragma unroll
  for (int g = 0; g < 16; ++g) {
    int c = g * 16 + n;
    int h = c >> 5, d = c & 31;
#pragma unroll
    for (int r = 0; r < 8; ++r) {
      int tok = m0 + mb + r;
      float v = acc[g][r];
      if (widx == 2) {
        Vt[((b * Hh + h) * Dd + d) * Nn + tok] = (bf16)v;
      } else {
        bf16* dst = (widx == 0) ? Q : K;
        dst[((b * Hh + h) * Nn + tok) * Dd + d] = (bf16)v;
      }
    }
  }
}

// ---------------------------------------------------------------------------
// Flash attention: grid = B*H*(N/128); 8 waves, wave owns 16 query rows.
// Per 32-key block: bias preloaded as WMMA C operand, 2 score WMMAs,
// online softmax (16-lane shfl_xor trees match C-fragment row striping),
// probabilities bounced through per-wave LDS (C-layout -> A-layout),
// 2 PV WMMAs. Output written bf16 into [b][n][h*32+d] for the final GEMM.
// ---------------------------------------------------------------------------
__global__ __launch_bounds__(256) void attn_kernel(
    const bf16* __restrict__ Q, const bf16* __restrict__ K,
    const bf16* __restrict__ Vt, const float* __restrict__ bias_t,
    bf16* __restrict__ attO) {
  __shared__ __align__(32) bf16 pstage[8 * 16 * 32];  // 8KB, 1KB per wave

  int blk = blockIdx.x;
  int qb  = blk & 7;
  int h   = (blk >> 3) & 7;
  int b   = blk >> 6;
  int w   = threadIdx.x >> 5;
  int q0  = qb * 128 + w * 16;

  int l  = lane_id();
  int n  = l & 15;
  int mb = (l & 16) ? 8 : 0;

  const bf16*  Qbase = Q  + ((b * Hh + h) * Nn + q0) * Dd;
  const bf16*  Kbh   = K  + (b * Hh + h) * Nn * Dd;
  const bf16*  Vbh   = Vt + (b * Hh + h) * Dd * Nn;
  const float* biasb = bias_t + b * Ss * Ss;

  v16bf qfrag = load_frag_a(Qbase, Dd);  // full head dim in one A fragment

  v8f o0 = {}, o1 = {};
  float mrow[8], lrow[8];
#pragma unroll
  for (int r = 0; r < 8; ++r) { mrow[r] = -3.0e38f; lrow[r] = 0.0f; }

  int qrow_s = (q0 & (Ss - 1)) + mb;  // bias row base (tiled, no intra-tile wrap)
  bf16* pblk = pstage + w * (16 * 32);

  for (int k0 = 0; k0 < Nn; k0 += 32) {
    int kc0 = (k0 & (Ss - 1)) + n;  // bias col (k0 step 32 => no wrap over +31)

    // bias tiles directly as WMMA C operands: score = Q*K^T (Q prescaled) + bias
    v8f c0, c1;
#pragma unroll
    for (int r = 0; r < 8; ++r) {
      const float* bp = biasb + (qrow_s + r) * Ss + kc0;
      c0[r] = bp[0];
      c1[r] = bp[16];
    }

    v16bf kf0 = load_frag_bT(Kbh + k0 * Dd, Dd);         // keys k0..k0+15
    v16bf kf1 = load_frag_bT(Kbh + (k0 + 16) * Dd, Dd);  // keys k0+16..k0+31
    v8f s0 = wmma_bf16(qfrag, kf0, c0);
    v8f s1 = wmma_bf16(qfrag, kf1, c1);

    // ---- online softmax over this 32-key block ----
    v8f p0, p1;
#pragma unroll
    for (int r = 0; r < 8; ++r) {
      float t = fmaxf(s0[r], s1[r]);
#pragma unroll
      for (int off = 1; off < 16; off <<= 1)
        t = fmaxf(t, __shfl_xor(t, off, 32));  // stays inside 16-lane half
      float mnew  = fmaxf(mrow[r], t);
      float alpha = __builtin_exp2f((mrow[r] - mnew) * LOG2E);
      float e0 = __builtin_exp2f((s0[r] - mnew) * LOG2E);
      float e1 = __builtin_exp2f((s1[r] - mnew) * LOG2E);
      float rs = e0 + e1;
#pragma unroll
      for (int off = 1; off < 16; off <<= 1)
        rs += __shfl_xor(rs, off, 32);
      lrow[r] = lrow[r] * alpha + rs;
      mrow[r] = mnew;
      o0[r] *= alpha;
      o1[r] *= alpha;
      p0[r] = e0;
      p1[r] = e1;
    }

    // C-layout -> A-layout bounce through per-wave LDS (no barrier needed:
    // LDS ops are wave-ordered; s_wait_dscnt fences the bf16 stores).
#pragma unroll
    for (int r = 0; r < 8; ++r) {
      pblk[(mb + r) * 32 + n]      = (bf16)p0[r];
      pblk[(mb + r) * 32 + 16 + n] = (bf16)p1[r];
    }
    asm volatile("s_wait_dscnt 0" ::: "memory");
    v16bf pfrag = load_frag_a(pblk, 32);

    v16bf vf0 = load_frag_bT(Vbh + k0, Nn);             // d = 0..15
    v16bf vf1 = load_frag_bT(Vbh + 16 * Nn + k0, Nn);   // d = 16..31
    o0 = wmma_bf16(pfrag, vf0, o0);
    o1 = wmma_bf16(pfrag, vf1, o1);
  }

  // normalize and scatter into [b][token][h*32 + d] (bf16)
#pragma unroll
  for (int r = 0; r < 8; ++r) {
    float rcp = 1.0f / lrow[r];
    int tok = q0 + mb + r;
    bf16* dst = attO + (b * Nn + tok) * Cc + h * Dd;
    dst[n]      = (bf16)(o0[r] * rcp);
    dst[16 + n] = (bf16)(o1[r] * rcp);
  }
}

// ---------------------------------------------------------------------------
// Output projection: [8192x256] bf16 @ Wproj^T(bf16) -> f32 out.
// ---------------------------------------------------------------------------
__global__ __launch_bounds__(256) void proj_gemm(
    const bf16* __restrict__ attO, const bf16* __restrict__ Wtp,
    float* __restrict__ out) {
  int w  = threadIdx.x >> 5;
  int m0 = blockIdx.x * 128 + w * 16;  // row over B*N = 8192

  v8f acc[16] = {};
  const bf16* Abase = attO + m0 * Cc;
  for (int kk = 0; kk < Cc; kk += 32) {
    v16bf a = load_frag_a(Abase + kk, Cc);
#pragma unroll
    for (int g = 0; g < 16; ++g) {
      v16bf bfrag = load_frag_bT(Wtp + (g * 16) * Cc + kk, Cc);
      acc[g] = wmma_bf16(a, bfrag, acc[g]);
    }
  }

  int l = lane_id();
  int n = l & 15, mb = (l & 16) ? 8 : 0;
#pragma unroll
  for (int g = 0; g < 16; ++g) {
#pragma unroll
    for (int r = 0; r < 8; ++r) {
      out[(m0 + mb + r) * Cc + g * 16 + n] = acc[g][r];
    }
  }
}

// ---------------------------------------------------------------------------
// Host launcher. Workspace layout (~21 MB, all 256B aligned):
//   xb 4MB | Wtq/Wtk/Wtv/Wtp 128KB each | Q 4MB | K 4MB | Vt 4MB | attO 4MB
// ---------------------------------------------------------------------------
extern "C" void kernel_launch(void* const* d_in, const int* in_sizes, int n_in,
                              void* d_out, int out_size, void* d_ws,
                              size_t ws_size, hipStream_t stream) {
  const float* x    = (const float*)d_in[0];
  const float* bias = (const float*)d_in[1];
  const float* Wq   = (const float*)d_in[2];
  const float* Wk   = (const float*)d_in[3];
  const float* Wv   = (const float*)d_in[4];
  const float* Wp   = (const float*)d_in[5];
  float* out = (float*)d_out;

  char* ws = (char*)d_ws;
  size_t off = 0;
  auto alloc = [&](size_t bytes) -> char* {
    char* p = ws + off;
    off += (bytes + 255) & ~(size_t)255;
    return p;
  };

  const size_t xElems = (size_t)Bsz * Nn * Cc;  // 2M
  bf16* xb   = (bf16*)alloc(xElems * 2);
  bf16* Wtq  = (bf16*)alloc((size_t)Cc * Cc * 2);
  bf16* Wtk  = (bf16*)alloc((size_t)Cc * Cc * 2);
  bf16* Wtv  = (bf16*)alloc((size_t)Cc * Cc * 2);
  bf16* Wtp  = (bf16*)alloc((size_t)Cc * Cc * 2);
  bf16* Qd   = (bf16*)alloc(xElems * 2);
  bf16* Kd   = (bf16*)alloc(xElems * 2);
  bf16* Vtd  = (bf16*)alloc(xElems * 2);
  bf16* attO = (bf16*)alloc(xElems * 2);

  const float scale = 0.17677669529663687f;  // D^{-1/2}, folded into Wq

  cvt_f32_to_bf16<<<2048, 256, 0, stream>>>(x, xb, (int)xElems);
  cvt_w_transpose<<<256, 256, 0, stream>>>(Wq, Wtq, scale);
  cvt_w_transpose<<<256, 256, 0, stream>>>(Wk, Wtk, 1.0f);
  cvt_w_transpose<<<256, 256, 0, stream>>>(Wv, Wtv, 1.0f);
  cvt_w_transpose<<<256, 256, 0, stream>>>(Wp, Wtp, 1.0f);

  qkv_gemm<<<Bsz * 8 * 3, 256, 0, stream>>>(xb, Wtq, Wtk, Wtv, Qd, Kd, Vtd);
  attn_kernel<<<Bsz * Hh * (Nn / 128), 256, 0, stream>>>(Qd, Kd, Vtd, bias, attO);
  proj_gemm<<<(Bsz * Nn) / 128, 256, 0, stream>>>(attO, Wtp, out);
}
